// Transformer_16432544875152
// MI455X (gfx1250) — matrix operand order repaired and checked by reading, not compiled, not run on hardware
//
#include <hip/hip_runtime.h>
#include <hip/hip_bf16.h>
#include <math.h>

// ---------------- model constants ----------------
#define BB   32
#define SS   512
#define DDIM 512
#define HH   8
#define LLAY 4
#define FF   1024
#define TDIM 2048
#define DHD  64
#define MTOK (BB * SS)           // 16384
#define ATT_SCALE 0.125f         // 1/sqrt(64)

typedef __bf16 bf16;
typedef __attribute__((ext_vector_type(16))) __bf16 v16bf;
typedef __attribute__((ext_vector_type(8)))  float  v8f;

static __device__ __forceinline__ bf16  f2b(float f) { return (bf16)f; }
static __device__ __forceinline__ float b2f(bf16 h)  { return (float)h; }

// ---------------- elementwise kernels ----------------
__global__ void k_f32_to_bf16(const float* __restrict__ src, bf16* __restrict__ dst, long n) {
  long i = (long)blockIdx.x * blockDim.x + threadIdx.x;
  long stride = (long)gridDim.x * blockDim.x;
  for (; i < n; i += stride) dst[i] = f2b(src[i]);
}

__global__ void k_time_embed(const float* __restrict__ timev, bf16* __restrict__ out) {
  int t = blockIdx.x * blockDim.x + threadIdx.x;
  if (t >= BB * DDIM) return;
  int b = t / DDIM, j = t % DDIM;
  const int half = DDIM / 2;
  int jj = (j < half) ? j : j - half;
  float freq = __expf((float)jj * (-logf(10000.0f) / (float)(half - 1)));
  float e = timev[b] * freq;
  out[t] = f2b((j < half) ? __sinf(e) : __cosf(e));
}

__global__ void k_gelu(const float* __restrict__ src, bf16* __restrict__ dst, int n) {
  int i = blockIdx.x * blockDim.x + threadIdx.x;
  if (i >= n) return;
  float x = src[i];
  dst[i] = f2b(0.5f * x * (1.0f + erff(x * 0.70710678118654752f)));
}

__global__ void k_silu(const float* __restrict__ src, bf16* __restrict__ dst, int n) {
  int i = blockIdx.x * blockDim.x + threadIdx.x;
  if (i >= n) return;
  float x = src[i];
  dst[i] = f2b(x / (1.0f + __expf(-x)));
}

// h1 (already FiLM'd, bf16, [rows, 2F]) -> a * relu(g)  [rows, F]
__global__ void k_reglu(const bf16* __restrict__ h1, bf16* __restrict__ out, int rows) {
  int i = blockIdx.x * blockDim.x + threadIdx.x;
  if (i >= rows * FF) return;
  int r = i / FF, n = i - r * FF;
  float a = b2f(h1[(long)r * (2 * FF) + n]);
  float g = b2f(h1[(long)r * (2 * FF) + FF + n]);
  out[i] = f2b(a * fmaxf(g, 0.0f));
}

// one wave per token; D = 512 -> 16 elems/lane; optional relu after LN
__global__ void k_layernorm(const float* __restrict__ x, long row_stride,
                            const float* __restrict__ g, const float* __restrict__ b,
                            bf16* __restrict__ out, int ntok, int do_relu) {
  int wv   = (blockIdx.x * blockDim.x + threadIdx.x) >> 5;
  int lane = threadIdx.x & 31;
  if (wv >= ntok) return;
  const float* row = x + (long)wv * row_stride;
  float v[16];
  float s = 0.f;
#pragma unroll
  for (int i = 0; i < 16; ++i) { v[i] = row[lane + i * 32]; s += v[i]; }
#pragma unroll
  for (int off = 16; off >= 1; off >>= 1) s += __shfl_xor(s, off, 32);
  float mean = s * (1.0f / DDIM);
  float q = 0.f;
#pragma unroll
  for (int i = 0; i < 16; ++i) { float d = v[i] - mean; q += d * d; }
#pragma unroll
  for (int off = 16; off >= 1; off >>= 1) q += __shfl_xor(q, off, 32);
  float rstd = rsqrtf(q * (1.0f / DDIM) + 1e-5f);
  bf16* orow = out + (long)wv * DDIM;
#pragma unroll
  for (int i = 0; i < 16; ++i) {
    int c = lane + i * 32;
    float y = (v[i] - mean) * rstd * g[c] + b[c];
    if (do_relu) y = fmaxf(y, 0.0f);
    orow[c] = f2b(y);
  }
}

// ---------------- WMMA GEMM:  C[M,N] = A[M,K] @ W[N,K]^T + bias ----------------
// Block tile: 64(M) x 128(N) x 64(K); 8 waves = 2(m) x 4(n); 32x32 output per wave
// (4 accumulators, 8 WMMAs per barrier window, fragments reused 2x each).
// modes: 0 -> bf16 out ; 1 -> f32 out = res + acc ; 2 -> bf16 out with FiLM(tm) ; 3 -> f32 out
#define LDT 72   // LDS row stride (bf16 elements), 144B (16B aligned)

__global__ __launch_bounds__(256) void k_gemm(
    const bf16* __restrict__ A, const bf16* __restrict__ W,
    const float* __restrict__ bias,
    int Mrows, int Ncols, int Kdim, int mode,
    float* __restrict__ outF, bf16* __restrict__ outB,
    const float* __restrict__ res,
    const float* __restrict__ film, int Sdiv)
{
  __shared__ __align__(16) bf16 As[64 * LDT];    //  64 x 64 tile
  __shared__ __align__(16) bf16 Ws[128 * LDT];   // 128 x 64 tile

  int tid  = threadIdx.x;
  int lane = tid & 31;
  int wave = tid >> 5;
  int n0 = blockIdx.x * 128;
  int m0 = blockIdx.y * 64;

  int mg = wave >> 2;          // 0..1 : 32-row group
  int ng = wave & 3;           // 0..3 : 32-col group

  v8f acc[2][2] = {{{}, {}}, {{}, {}}};

  int mr  = lane & 15;
  int kA  = (lane >> 4) * 8;    // interleaved K base (unused; A frags use kB)
  int kB  = (lane >> 4) * 16;   // B frag linear K base
  int rbase = (lane >> 4) * 8;

  for (int kk = 0; kk < Kdim; kk += 64) {
    // cooperative load: A 64x64 (512 uint4), W 128x64 (1024 uint4)
#pragma unroll
    for (int u = 0; u < 2; ++u) {
      int f = tid + u * 256;
      int row = f >> 3;
      int col = (f & 7) * 8;
      uint4 da = make_uint4(0u, 0u, 0u, 0u);
      int gm = m0 + row;
      if (gm < Mrows) da = *(const uint4*)(A + (long)gm * Kdim + kk + col);
      *(uint4*)(&As[row * LDT + col]) = da;
    }
#pragma unroll
    for (int u = 0; u < 4; ++u) {
      int f = tid + u * 256;
      int row = f >> 3;
      int col = (f & 7) * 8;
      *(uint4*)(&Ws[row * LDT + col]) = *(const uint4*)(W + (long)(n0 + row) * Kdim + kk + col);
    }
    if (kk + 64 < Kdim) {   // prefetch next K tile (global_prefetch_b8)
      __builtin_prefetch(W + (long)(n0 + (tid >> 1)) * Kdim + kk + 64 + (tid & 1) * 32, 0, 1);
      __builtin_prefetch(A + (long)(m0 + (tid & 63)) * Kdim + kk + 64 + (tid >> 6) * 16, 0, 1);
    }
    __syncthreads();

#pragma unroll
    for (int c = 0; c < 2; ++c) {
      int ko = c * 32;
      v16bf a0, a1, b0, b1;
#pragma unroll
      for (int e = 0; e < 16; ++e) {
        int ka = ko + kB + e;
        a0[e] = As[(mg * 32 + mr) * LDT + ka];
        a1[e] = As[(mg * 32 + 16 + mr) * LDT + ka];
        int kb = ko + kB + e;
        b0[e] = Ws[(ng * 32 + mr) * LDT + kb];
        b1[e] = Ws[(ng * 32 + 16 + mr) * LDT + kb];
      }
      acc[0][0] = __builtin_amdgcn_wmma_f32_16x16x32_bf16(false, a0, false, b0, (short)0, acc[0][0], false, false);
      acc[0][1] = __builtin_amdgcn_wmma_f32_16x16x32_bf16(false, a0, false, b1, (short)0, acc[0][1], false, false);
      acc[1][0] = __builtin_amdgcn_wmma_f32_16x16x32_bf16(false, a1, false, b0, (short)0, acc[1][0], false, false);
      acc[1][1] = __builtin_amdgcn_wmma_f32_16x16x32_bf16(false, a1, false, b1, (short)0, acc[1][1], false, false);
    }
    __syncthreads();
  }

  // epilogue: 4 tiles of 16x16; element r <-> row rbase+r, col mr
#pragma unroll
  for (int ti = 0; ti < 2; ++ti) {
#pragma unroll
    for (int tj = 0; tj < 2; ++tj) {
      int cN = n0 + ng * 32 + tj * 16 + mr;
#pragma unroll
      for (int r = 0; r < 8; ++r) {
        int gm = m0 + mg * 32 + ti * 16 + rbase + r;
        if (gm >= Mrows) continue;
        float v = acc[ti][tj][r] + (bias ? bias[cN] : 0.0f);
        long idx = (long)gm * Ncols + cN;
        if (mode == 0) {
          outB[idx] = f2b(v);
        } else if (mode == 1) {
          outF[idx] = res[idx] + v;
        } else if (mode == 2) {
          int bi = gm / Sdiv;
          const float* fp = film + (long)bi * (2 * Ncols);
          outB[idx] = f2b(v * (fp[cN] + 1.0f) + fp[Ncols + cN]);
        } else {
          outF[idx] = v;
        }
      }
    }
  }
}

// ---------------- flash attention ----------------
// grid: B*H*4 blocks, 256 threads (8 waves); wave handles 16 q rows; K/V in LDS
#define KLD 72   // 32x64 tile row stride (bf16), 144B (16B aligned)

__global__ __launch_bounds__(256) void k_attn(
    const bf16* __restrict__ Q, const bf16* __restrict__ Km,
    const bf16* __restrict__ Vm, bf16* __restrict__ O)
{
  __shared__ __align__(16) bf16 Kt[32 * KLD];
  __shared__ __align__(16) bf16 Vt[32 * KLD];
  __shared__ __align__(16) bf16 Pw[8 * 16 * 40];   // per-wave 16x32 P scratch

  int tid  = threadIdx.x;
  int lane = tid & 31;
  int wave = tid >> 5;
  int qb = blockIdx.x & 3;
  int bh = blockIdx.x >> 2;
  int h  = bh % HH, b = bh / HH;
  int q0 = qb * 128 + wave * 16;

  const long base = ((long)b * SS) * DDIM + h * DHD;

  int mr    = lane & 15;
  int kA    = (lane >> 4) * 8;
  int kB16  = (lane >> 4) * 16;
  int rbase = (lane >> 4) * 8;

  // Q fragments (pre-scaled by 1/sqrt(dh))
  v16bf aq0, aq1;
  const bf16* qrow = Q + base + (long)(q0 + mr) * DDIM;
#pragma unroll
  for (int e = 0; e < 16; ++e) {
    int k = kB16 + e;
    aq0[e] = f2b(ATT_SCALE * b2f(qrow[k]));
    aq1[e] = f2b(ATT_SCALE * b2f(qrow[k + 32]));
  }

  float smax[8], ssum[8];
  v8f o0 = {}, o1 = {}, o2 = {}, o3 = {};
#pragma unroll
  for (int r = 0; r < 8; ++r) { smax[r] = -1e30f; ssum[r] = 0.0f; }

  int ldr = tid >> 3;          // 0..31
  int ldc = (tid & 7) * 8;     // 0..56

  for (int kt = 0; kt < SS; kt += 32) {
    *(uint4*)(&Kt[ldr * KLD + ldc]) = *(const uint4*)(Km + base + (long)(kt + ldr) * DDIM + ldc);
    *(uint4*)(&Vt[ldr * KLD + ldc]) = *(const uint4*)(Vm + base + (long)(kt + ldr) * DDIM + ldc);
    __syncthreads();

    // S = Q @ K^T : two 16x16 tiles (keys 0-15, 16-31), K-dim = dh 64 split 32+32
    v8f s0 = {}, s1 = {};
    {
      v16bf bk;
#pragma unroll
      for (int e = 0; e < 16; ++e) bk[e] = Kt[mr * KLD + kB16 + e];
      s0 = __builtin_amdgcn_wmma_f32_16x16x32_bf16(false, aq0, false, bk, (short)0, s0, false, false);
#pragma unroll
      for (int e = 0; e < 16; ++e) bk[e] = Kt[mr * KLD + 32 + kB16 + e];
      s0 = __builtin_amdgcn_wmma_f32_16x16x32_bf16(false, aq1, false, bk, (short)0, s0, false, false);
#pragma unroll
      for (int e = 0; e < 16; ++e) bk[e] = Kt[(16 + mr) * KLD + kB16 + e];
      s1 = __builtin_amdgcn_wmma_f32_16x16x32_bf16(false, aq0, false, bk, (short)0, s1, false, false);
#pragma unroll
      for (int e = 0; e < 16; ++e) bk[e] = Kt[(16 + mr) * KLD + 32 + kB16 + e];
      s1 = __builtin_amdgcn_wmma_f32_16x16x32_bf16(false, aq1, false, bk, (short)0, s1, false, false);
    }

    // online softmax update (row reductions across the 16-lane half-groups)
    float corr[8];
#pragma unroll
    for (int r = 0; r < 8; ++r) {
      float mx = fmaxf(s0[r], s1[r]);
#pragma unroll
      for (int off = 1; off < 16; off <<= 1) mx = fmaxf(mx, __shfl_xor(mx, off, 32));
      float nm = fmaxf(smax[r], mx);
      corr[r] = __expf(smax[r] - nm);
      smax[r] = nm;
      float p0 = __expf(s0[r] - nm);
      float p1 = __expf(s1[r] - nm);
      s0[r] = p0; s1[r] = p1;
      float rs = p0 + p1;
#pragma unroll
      for (int off = 1; off < 16; off <<= 1) rs += __shfl_xor(rs, off, 32);
      ssum[r] = ssum[r] * corr[r] + rs;
      o0[r] *= corr[r]; o1[r] *= corr[r]; o2[r] *= corr[r]; o3[r] *= corr[r];
    }

    // re-layout P (C format -> A format) via per-wave LDS scratch
    bf16* P = &Pw[wave * 16 * 40];
#pragma unroll
    for (int r = 0; r < 8; ++r) {
      P[(rbase + r) * 40 + mr]      = f2b(s0[r]);
      P[(rbase + r) * 40 + 16 + mr] = f2b(s1[r]);
    }
    asm volatile("s_wait_dscnt 0" ::: "memory");
    v16bf pa;
#pragma unroll
    for (int e = 0; e < 16; ++e) {
      int k = kB16 + e;
      pa[e] = P[mr * 40 + k];
    }

    // O += P(16x32) @ V(32x64)
    v16bf bv;
#pragma unroll
    for (int e = 0; e < 16; ++e) bv[e] = Vt[(kB16 + e) * KLD + mr];
    o0 = __builtin_amdgcn_wmma_f32_16x16x32_bf16(false, pa, false, bv, (short)0, o0, false, false);
#pragma unroll
    for (int e = 0; e < 16; ++e) bv[e] = Vt[(kB16 + e) * KLD + 16 + mr];
    o1 = __builtin_amdgcn_wmma_f32_16x16x32_bf16(false, pa, false, bv, (short)0, o1, false, false);
#pragma unroll
    for (int e = 0; e < 16; ++e) bv[e] = Vt[(kB16 + e) * KLD + 32 + mr];
    o2 = __builtin_amdgcn_wmma_f32_16x16x32_bf16(false, pa, false, bv, (short)0, o2, false, false);
#pragma unroll
    for (int e = 0; e < 16; ++e) bv[e] = Vt[(kB16 + e) * KLD + 48 + mr];
    o3 = __builtin_amdgcn_wmma_f32_16x16x32_bf16(false, pa, false, bv, (short)0, o3, false, false);
    __syncthreads();
  }

#pragma unroll
  for (int r = 0; r < 8; ++r) {
    float inv = 1.0f / ssum[r];
    long row = base + (long)(q0 + rbase + r) * DDIM;
    O[row +      mr] = f2b(o0[r] * inv);
    O[row + 16 + mr] = f2b(o1[r] * inv);
    O[row + 32 + mr] = f2b(o2[r] * inv);
    O[row + 48 + mr] = f2b(o3[r] * inv);
  }
}

// ---------------- host orchestration ----------------
extern "C" void kernel_launch(void* const* d_in, const int* in_sizes, int n_in,
                              void* d_out, int out_size, void* d_ws, size_t ws_size,
                              hipStream_t stream)
{
  const float* x_in    = (const float*)d_in[0];
  const float* timev   = (const float*)d_in[1];
  const float* time_w1 = (const float*)d_in[2];
  const float* time_b1 = (const float*)d_in[3];
  const float* time_w2 = (const float*)d_in[4];
  const float* time_b2 = (const float*)d_in[5];
  const float* Wq = (const float*)d_in[6];   const float* bq = (const float*)d_in[7];
  const float* Wk = (const float*)d_in[8];   const float* bk = (const float*)d_in[9];
  const float* Wv = (const float*)d_in[10];  const float* bv = (const float*)d_in[11];
  const float* Wo = (const float*)d_in[12];  const float* bo = (const float*)d_in[13];
  const float* attn_g = (const float*)d_in[14]; const float* attn_b = (const float*)d_in[15];
  const float* ffn_g  = (const float*)d_in[16]; const float* ffn_b  = (const float*)d_in[17];
  const float* W1 = (const float*)d_in[18];  const float* b1 = (const float*)d_in[19];
  const float* W2 = (const float*)d_in[20];  const float* b2 = (const float*)d_in[21];
  const float* Wt = (const float*)d_in[22];  const float* bt = (const float*)d_in[23];
  const float* head_g = (const float*)d_in[24]; const float* head_b = (const float*)d_in[25];
  const float* Wh = (const float*)d_in[26];  const float* bh = (const float*)d_in[27];

  char* ws = (char*)d_ws;
  size_t off = 0;
  auto alloc = [&](size_t bytes) -> void* {
    void* p = ws + off;
    off = (off + bytes + 255) & ~(size_t)255;
    return p;
  };

  bf16* tw1b = (bf16*)alloc((size_t)TDIM * DDIM * 2);
  bf16* tw2b = (bf16*)alloc((size_t)TDIM * TDIM * 2);
  bf16* wqb  = (bf16*)alloc((size_t)LLAY * DDIM * DDIM * 2);
  bf16* wkb  = (bf16*)alloc((size_t)LLAY * DDIM * DDIM * 2);
  bf16* wvb  = (bf16*)alloc((size_t)LLAY * DDIM * DDIM * 2);
  bf16* wob  = (bf16*)alloc((size_t)LLAY * DDIM * DDIM * 2);
  bf16* w1b  = (bf16*)alloc((size_t)LLAY * 2 * FF * DDIM * 2);
  bf16* w2b  = (bf16*)alloc((size_t)LLAY * DDIM * FF * 2);
  bf16* wtb  = (bf16*)alloc((size_t)LLAY * 4 * FF * TDIM * 2);
  bf16* whb  = (bf16*)alloc((size_t)DDIM * DDIM * 2);

  float* x   = (float*)alloc((size_t)MTOK * DDIM * 4);
  bf16* hb   = (bf16*)alloc((size_t)MTOK * DDIM * 2);
  bf16* qbuf = (bf16*)alloc((size_t)MTOK * DDIM * 2);
  bf16* kbuf = (bf16*)alloc((size_t)MTOK * DDIM * 2);
  bf16* vbuf = (bf16*)alloc((size_t)MTOK * DDIM * 2);
  bf16* obuf = (bf16*)alloc((size_t)MTOK * DDIM * 2);
  bf16* h1b  = (bf16*)alloc((size_t)MTOK * 2 * FF * 2);
  bf16* g2b  = (bf16*)alloc((size_t)MTOK * FF * 2);

  bf16*  embb  = (bf16*)alloc((size_t)BB * DDIM * 2);
  float* t1f   = (float*)alloc((size_t)BB * TDIM * 4);
  bf16*  t1gb  = (bf16*)alloc((size_t)BB * TDIM * 2);
  float* tembf = (float*)alloc((size_t)BB * TDIM * 4);
  bf16*  stb   = (bf16*)alloc((size_t)BB * TDIM * 2);
  float* tmf   = (float*)alloc((size_t)BB * 4 * FF * 4);
  bf16*  zb    = (bf16*)alloc((size_t)BB * DDIM * 2);

  auto conv = [&](const float* s, bf16* d, long n) {
    long blocks = (n + 255) / 256;
    if (blocks > 8192) blocks = 8192;
    k_f32_to_bf16<<<(int)blocks, 256, 0, stream>>>(s, d, n);
  };
  auto gemm = [&](const bf16* A, const bf16* Wm, const float* bias,
                  int Mr, int Nc, int Kd, int mode,
                  float* oF, bf16* oB, const float* res, const float* film) {
    dim3 grid(Nc / 128, (Mr + 63) / 64);
    k_gemm<<<grid, 256, 0, stream>>>(A, Wm, bias, Mr, Nc, Kd, mode, oF, oB, res, film, SS);
  };

  // residual stream
  hipMemcpyAsync(x, x_in, (size_t)MTOK * DDIM * 4, hipMemcpyDeviceToDevice, stream);

  // weight conversions (bf16)
  conv(time_w1, tw1b, (long)TDIM * DDIM);
  conv(time_w2, tw2b, (long)TDIM * TDIM);
  conv(Wq, wqb, (long)LLAY * DDIM * DDIM);
  conv(Wk, wkb, (long)LLAY * DDIM * DDIM);
  conv(Wv, wvb, (long)LLAY * DDIM * DDIM);
  conv(Wo, wob, (long)LLAY * DDIM * DDIM);
  conv(W1, w1b, (long)LLAY * 2 * FF * DDIM);
  conv(W2, w2b, (long)LLAY * DDIM * FF);
  conv(Wt, wtb, (long)LLAY * 4 * FF * TDIM);
  conv(Wh, whb, (long)DDIM * DDIM);

  // time embedding + time MLP
  k_time_embed<<<(BB * DDIM + 255) / 256, 256, 0, stream>>>(timev, embb);
  gemm(embb, tw1b, time_b1, BB, TDIM, DDIM, 3, t1f, nullptr, nullptr, nullptr);
  k_gelu<<<(BB * TDIM + 255) / 256, 256, 0, stream>>>(t1f, t1gb, BB * TDIM);
  gemm(t1gb, tw2b, time_b2, BB, TDIM, TDIM, 3, tembf, nullptr, nullptr, nullptr);
  k_silu<<<(BB * TDIM + 255) / 256, 256, 0, stream>>>(tembf, stb, BB * TDIM);

  const int ln_grid = (MTOK + 7) / 8;   // 8 tokens per 256-thread block
  for (int i = 0; i < LLAY; ++i) {
    // pre-attn norm (layer 0 skips norm)
    if (i == 0) conv(x, hb, (long)MTOK * DDIM);
    else k_layernorm<<<ln_grid, 256, 0, stream>>>(x, DDIM, attn_g + i * DDIM, attn_b + i * DDIM,
                                                  hb, MTOK, 0);
    // QKV projections
    gemm(hb, wqb + (size_t)i * DDIM * DDIM, bq + i * DDIM, MTOK, DDIM, DDIM, 0, nullptr, qbuf, nullptr, nullptr);
    gemm(hb, wkb + (size_t)i * DDIM * DDIM, bk + i * DDIM, MTOK, DDIM, DDIM, 0, nullptr, kbuf, nullptr, nullptr);
    gemm(hb, wvb + (size_t)i * DDIM * DDIM, bv + i * DDIM, MTOK, DDIM, DDIM, 0, nullptr, vbuf, nullptr, nullptr);
    // attention
    k_attn<<<BB * HH * 4, 256, 0, stream>>>(qbuf, kbuf, vbuf, obuf);
    // output projection + residual
    gemm(obuf, wob + (size_t)i * DDIM * DDIM, bo + i * DDIM, MTOK, DDIM, DDIM, 1, x, nullptr, x, nullptr);
    // FFN
    k_layernorm<<<ln_grid, 256, 0, stream>>>(x, DDIM, ffn_g + i * DDIM, ffn_b + i * DDIM, hb, MTOK, 0);
    gemm(stb, wtb + (size_t)i * 4 * FF * TDIM, bt + i * 4 * FF, BB, 4 * FF, TDIM, 3, tmf, nullptr, nullptr, nullptr);
    gemm(hb, w1b + (size_t)i * 2 * FF * DDIM, b1 + i * 2 * FF, MTOK, 2 * FF, DDIM, 2, nullptr, h1b, nullptr, tmf);
    k_reglu<<<(MTOK * FF + 255) / 256, 256, 0, stream>>>(h1b, g2b, MTOK);
    gemm(g2b, w2b + (size_t)i * DDIM * FF, b2 + i * DDIM, MTOK, DDIM, FF, 1, x, nullptr, x, nullptr);
  }

  // head: last token -> LN -> ReLU -> Linear
  k_layernorm<<<(BB + 7) / 8, 256, 0, stream>>>(x + (size_t)(SS - 1) * DDIM, (long)SS * DDIM,
                                                head_g, head_b, zb, BB, 1);
  gemm(zb, whb, bh, BB, DDIM, DDIM, 3, (float*)d_out, nullptr, nullptr, nullptr);
}
